// LanguageModelGroupedQuaryAttention_67087389163537
// MI455X (gfx1250) — compile-verified, matrix operand order checked by reading, not run on hardware
//
#include <hip/hip_runtime.h>
#include <stdint.h>

#define T_SEQ 2048
#define E_DIM 1024
#define NH 16
#define NKVH 4
#define HDIM 64
#define NTOK 4096  // B*T

typedef __attribute__((ext_vector_type(16))) __bf16 v16bf;
typedef __attribute__((ext_vector_type(8)))  __bf16 v8bf;
typedef __attribute__((ext_vector_type(8)))  float  v8f;

__device__ __forceinline__ unsigned short f2bf(float f) {
  unsigned u = __float_as_uint(f);
  unsigned r = u + 0x7FFFu + ((u >> 16) & 1u);
  return (unsigned short)(r >> 16);
}

__device__ __forceinline__ v8f zero8() {
  v8f z;
#pragma unroll
  for (int i = 0; i < 8; ++i) z[i] = 0.0f;
  return z;
}

// LDS byte offset of a generic shared pointer (low 32 bits of the flat address).
__device__ __forceinline__ unsigned lds_addr(const void* p) {
  return (unsigned)(unsigned long long)(uintptr_t)p;
}

// CDNA5 async global->LDS copy, 16 bytes per lane (ASYNCcnt-tracked).
__device__ __forceinline__ void async_copy16(unsigned lds_off, const void* gptr) {
  asm volatile("global_load_async_to_lds_b128 %0, %1, off"
               :: "v"(lds_off), "v"((unsigned long long)gptr)
               : "memory");
}
__device__ __forceinline__ void async_wait0() {
  asm volatile("s_wait_asynccnt 0x0" ::: "memory");
}

// A/B fragment per ISA 7.12.2 (16-bit, 16x32): lane = {row, K-half}; each lane
// holds two contiguous 8-element K segments: K = 8*half.. and 16+8*half..
__device__ __forceinline__ v16bf load_frag(const unsigned short* base, int hf) {
  v8bf lo = *(const v8bf*)(base + 8 * hf);
  v8bf hi = *(const v8bf*)(base + 16 + 8 * hf);
  v16bf r;
#pragma unroll
  for (int i = 0; i < 8; ++i) { r[i] = lo[i]; r[i + 8] = hi[i]; }
  return r;
}

__device__ __forceinline__ v8f wmma_bf16(v16bf a, v16bf b, v8f c) {
  return __builtin_amdgcn_wmma_f32_16x16x32_bf16(false, a, false, b, (short)0, c,
                                                 false, false);
}

// ---------------- f32 -> bf16 convert ----------------
__global__ void cvt_bf16(const float* __restrict__ in, unsigned short* __restrict__ out,
                         int n) {
  int i = blockIdx.x * blockDim.x + threadIdx.x;
  for (; i < n; i += gridDim.x * blockDim.x) out[i] = f2bf(in[i]);
}

// ---------------- RoPE (+ bf16 convert) ----------------
__global__ void rope_cvt(const float* __restrict__ in, const float* __restrict__ cosb,
                         const float* __restrict__ sinb, unsigned short* __restrict__ out,
                         int nH, int applyRope) {
  int idx = blockIdx.x * blockDim.x + threadIdx.x;
  int total = NTOK * nH * 32;
  if (idx >= total) return;
  int d = idx & 31;
  int h = (idx >> 5) % nH;
  int t = idx / (32 * nH);  // global token (b*T + tt); cos/sin indexed the same way
  int row = t * nH * HDIM + h * HDIM;
  float a = in[row + d], b = in[row + d + 32];
  float o1, o2;
  if (applyRope) {
    float c1 = cosb[t * HDIM + d],      s1 = sinb[t * HDIM + d];
    float c2 = cosb[t * HDIM + 32 + d], s2 = sinb[t * HDIM + 32 + d];
    o1 = a * c1 - b * s1;
    o2 = b * c2 + a * s2;
  } else {
    o1 = a; o2 = b;
  }
  out[row + d] = f2bf(o1);
  out[row + d + 32] = f2bf(o2);
}

// ---------------- bf16 WMMA GEMM: C[m,n] = sum_k A[m,k]*B[n,k] ----------------
// A: (M x K) bf16 row-major, B: (N x K) bf16 row-major, C: (M x N) f32.
// Block tile 128x64, 256 threads = 8 waves in a 4x2 grid; 32x32 per wave.
// Tiles staged with global_load_async_to_lds_b128 (ASYNCcnt path).
__global__ __launch_bounds__(256) void gemm_bf16_wmma(
    const unsigned short* __restrict__ A, const unsigned short* __restrict__ Bw,
    float* __restrict__ C, int K, int N) {
  __shared__ unsigned short Asm[128 * 40];
  __shared__ unsigned short Bsm[64 * 40];
  const int tid = threadIdx.x;
  const int lane = tid & 31, wid = tid >> 5;
  const int wr = wid & 3, wc = wid >> 2;
  const int bm = blockIdx.y * 128, bn = blockIdx.x * 64;
  const int ln = lane & 15, hf = lane >> 4;

  v8f acc[2][2];
#pragma unroll
  for (int i = 0; i < 2; ++i)
#pragma unroll
    for (int j = 0; j < 2; ++j) acc[i][j] = zero8();

  const int arow = tid >> 1, aseg = (tid & 1) * 16;  // 128 rows x 32 cols
  const int brow = tid >> 2, bseg = (tid & 3) * 8;   // 64 rows x 32 cols
  const unsigned ldsA = lds_addr(Asm + arow * 40 + aseg);
  const unsigned ldsB = lds_addr(Bsm + brow * 40 + bseg);

  for (int k0 = 0; k0 < K; k0 += 32) {
    const unsigned short* ga = A + (size_t)(bm + arow) * K + k0 + aseg;
    const unsigned short* gb = Bw + (size_t)(bn + brow) * K + k0 + bseg;
    async_copy16(ldsA, ga);
    async_copy16(ldsA + 16, ga + 8);
    async_copy16(ldsB, gb);
    if (k0 + 32 < K) {  // global_prefetch_b8 for the next K slab
      __builtin_prefetch(ga + 32, 0, 3);
      __builtin_prefetch(gb + 32, 0, 3);
    }
    async_wait0();
    __syncthreads();

    v16bf af[2], bfr[2];
#pragma unroll
    for (int mi = 0; mi < 2; ++mi)
      af[mi] = load_frag(Asm + (wr * 32 + mi * 16 + ln) * 40, hf);
#pragma unroll
    for (int ni = 0; ni < 2; ++ni)
      bfr[ni] = load_frag(Bsm + (wc * 32 + ni * 16 + ln) * 40, hf);
#pragma unroll
    for (int mi = 0; mi < 2; ++mi)
#pragma unroll
      for (int ni = 0; ni < 2; ++ni)
        acc[mi][ni] = wmma_bf16(af[mi], bfr[ni], acc[mi][ni]);
    __syncthreads();
  }

  // C layout: VGPR r -> M = r + 8*(lane>=16), N = lane&15
#pragma unroll
  for (int mi = 0; mi < 2; ++mi)
#pragma unroll
    for (int ni = 0; ni < 2; ++ni) {
      float* cp = C + (size_t)(bm + wr * 32 + mi * 16 + 8 * hf) * N +
                  (bn + wc * 32 + ni * 16 + ln);
#pragma unroll
      for (int r = 0; r < 8; ++r) cp[r * N] = acc[mi][ni][r];
    }
}

// ---------------- Flash attention (GQA, causal) ----------------
// Grid: (T/64, H, B). Block: 128 threads = 4 waves; each wave owns 16 query rows.
__global__ __launch_bounds__(128) void flash_gqa(
    const unsigned short* __restrict__ Q, const unsigned short* __restrict__ Kc,
    const unsigned short* __restrict__ Vc, unsigned short* __restrict__ Y) {
  __shared__ unsigned short Ksm[64 * 72];      // keys x HD (row-major)
  __shared__ unsigned short Vtsm[64 * 72];     // HD x keys (transposed)
  __shared__ unsigned short Psm[4][16 * 72];   // per-wave probs, rows x keys
  const int tid = threadIdx.x;
  const int lane = tid & 31, wid = tid >> 5;
  const int ln = lane & 15, hf = lane >> 4;
  const int qtile = blockIdx.x, h = blockIdx.y, b = blockIdx.z;
  const int qbase = qtile * 64;
  const int kvh = h >> 2;  // G = 4
  const size_t tok0 = (size_t)b * T_SEQ;

  // Q A-fragments straight from global (rows are contiguous in head-dim)
  const unsigned short* qrow =
      Q + (tok0 + qbase + wid * 16 + ln) * E_DIM + h * HDIM;
  v16bf qa0 = load_frag(qrow, hf);
  v16bf qa1 = load_frag(qrow + 32, hf);

  float rmax[8], rsum[8];
  v8f o[4];
#pragma unroll
  for (int r = 0; r < 8; ++r) { rmax[r] = -1e30f; rsum[r] = 0.0f; }
#pragma unroll
  for (int ni = 0; ni < 4; ++ni) o[ni] = zero8();

  const int krow = tid >> 1, koff = (tid & 1) * 32;
  const unsigned ldsK = lds_addr(Ksm + krow * 72 + koff);

  for (int kb = 0; kb <= qbase + 63; kb += 64) {
    __syncthreads();  // previous iteration's LDS reads complete
    {
      // K tile: async global->LDS (straight copy)
      const unsigned short* gk =
          Kc + (tok0 + kb + krow) * (NKVH * HDIM) + kvh * HDIM + koff;
#pragma unroll
      for (int i = 0; i < 4; ++i) async_copy16(ldsK + i * 16, gk + i * 8);
      // V tile: transposed scatter into LDS
#pragma unroll
      for (int i = 0; i < 32; ++i) {
        int idx = tid + 128 * i;
        int key = idx >> 6, hd = idx & 63;
        Vtsm[hd * 72 + key] = Vc[(tok0 + kb + key) * (NKVH * HDIM) + kvh * HDIM + hd];
      }
    }
    async_wait0();
    __syncthreads();

    // S = Q @ K^T over HD=64 (two 32-K WMMA steps), 4 key tiles of 16
    v8f s[4];
#pragma unroll
    for (int ni = 0; ni < 4; ++ni) {
      v16bf bk0 = load_frag(Ksm + (ni * 16 + ln) * 72, hf);
      v16bf bk1 = load_frag(Ksm + (ni * 16 + ln) * 72 + 32, hf);
      s[ni] = zero8();
      s[ni] = wmma_bf16(qa0, bk0, s[ni]);
      s[ni] = wmma_bf16(qa1, bk1, s[ni]);
    }
    // scale + causal mask
#pragma unroll
    for (int ni = 0; ni < 4; ++ni)
#pragma unroll
      for (int r = 0; r < 8; ++r) {
        int qg = qbase + wid * 16 + r + 8 * hf;
        int kg = kb + ni * 16 + ln;
        float v = s[ni][r] * 0.125f;  // 1/sqrt(64)
        s[ni][r] = (kg > qg) ? -1e30f : v;
      }
    // online softmax; row lives in a 16-lane half, reduce via shfl_xor 1..8
#pragma unroll
    for (int r = 0; r < 8; ++r) {
      float lm = fmaxf(fmaxf(s[0][r], s[1][r]), fmaxf(s[2][r], s[3][r]));
#pragma unroll
      for (int m = 1; m < 16; m <<= 1) lm = fmaxf(lm, __shfl_xor(lm, m, 32));
      float nm = fmaxf(rmax[r], lm);
      float corr = __expf(rmax[r] - nm);
      float lsum = 0.0f;
#pragma unroll
      for (int ni = 0; ni < 4; ++ni) {
        float p = __expf(s[ni][r] - nm);
        s[ni][r] = p;
        lsum += p;
      }
#pragma unroll
      for (int m = 1; m < 16; m <<= 1) lsum += __shfl_xor(lsum, m, 32);
      rsum[r] = rsum[r] * corr + lsum;
      rmax[r] = nm;
#pragma unroll
      for (int ni = 0; ni < 4; ++ni) o[ni][r] *= corr;
    }
    // P (C-layout) -> LDS -> A-layout for the attn@V WMMA
#pragma unroll
    for (int ni = 0; ni < 4; ++ni)
#pragma unroll
      for (int r = 0; r < 8; ++r)
        Psm[wid][(r + 8 * hf) * 72 + ni * 16 + ln] = f2bf(s[ni][r]);
    __syncthreads();

    v16bf pa0 = load_frag(&Psm[wid][ln * 72], hf);
    v16bf pa1 = load_frag(&Psm[wid][ln * 72 + 32], hf);
#pragma unroll
    for (int ni = 0; ni < 4; ++ni) {
      v16bf vb0 = load_frag(Vtsm + (ni * 16 + ln) * 72, hf);
      v16bf vb1 = load_frag(Vtsm + (ni * 16 + ln) * 72 + 32, hf);
      o[ni] = wmma_bf16(pa0, vb0, o[ni]);
      o[ni] = wmma_bf16(pa1, vb1, o[ni]);
    }
  }

  // normalize and store y in (b,t,h*HD+hd) layout (bf16) for the final GEMM
#pragma unroll
  for (int ni = 0; ni < 4; ++ni)
#pragma unroll
    for (int r = 0; r < 8; ++r) {
      size_t t = tok0 + qbase + wid * 16 + r + 8 * hf;
      Y[t * E_DIM + h * HDIM + ni * 16 + ln] = f2bf(o[ni][r] / rsum[r]);
    }
}

extern "C" void kernel_launch(void* const* d_in, const int* in_sizes, int n_in,
                              void* d_out, int out_size, void* d_ws, size_t ws_size,
                              hipStream_t stream) {
  const float* x    = (const float*)d_in[0];
  const float* cosb = (const float*)d_in[1];
  const float* sinb = (const float*)d_in[2];
  const float* Wq   = (const float*)d_in[3];
  const float* Wk   = (const float*)d_in[4];
  const float* Wv   = (const float*)d_in[5];
  const float* Wo   = (const float*)d_in[6];

  char* ws = (char*)d_ws;
  size_t off = 0;
  auto alloc = [&](size_t bytes) -> void* {
    void* p = ws + off;
    off += (bytes + 255) & ~(size_t)255;
    return p;
  };

  unsigned short* xb  = (unsigned short*)alloc((size_t)NTOK * E_DIM * 2);
  unsigned short* Wqb = (unsigned short*)alloc((size_t)E_DIM * E_DIM * 2);
  unsigned short* Wkb = (unsigned short*)alloc((size_t)NKVH * HDIM * E_DIM * 2);
  unsigned short* Wvb = (unsigned short*)alloc((size_t)NKVH * HDIM * E_DIM * 2);
  unsigned short* Wob = (unsigned short*)alloc((size_t)E_DIM * E_DIM * 2);
  float* Qf = (float*)alloc((size_t)NTOK * E_DIM * 4);
  float* Kf = (float*)alloc((size_t)NTOK * NKVH * HDIM * 4);
  float* Vf = (float*)alloc((size_t)NTOK * NKVH * HDIM * 4);
  unsigned short* Qr = (unsigned short*)alloc((size_t)NTOK * E_DIM * 2);
  unsigned short* Kr = (unsigned short*)alloc((size_t)NTOK * NKVH * HDIM * 2);
  unsigned short* Vb = (unsigned short*)alloc((size_t)NTOK * NKVH * HDIM * 2);
  unsigned short* Yb = (unsigned short*)alloc((size_t)NTOK * E_DIM * 2);

  const int nx = NTOK * E_DIM;            // 4,194,304
  const int nwq = E_DIM * E_DIM;          // 1,048,576
  const int nwk = NKVH * HDIM * E_DIM;    // 262,144

  cvt_bf16<<<(nx + 255) / 256, 256, 0, stream>>>(x, xb, nx);
  cvt_bf16<<<(nwq + 255) / 256, 256, 0, stream>>>(Wq, Wqb, nwq);
  cvt_bf16<<<(nwk + 255) / 256, 256, 0, stream>>>(Wk, Wkb, nwk);
  cvt_bf16<<<(nwk + 255) / 256, 256, 0, stream>>>(Wv, Wvb, nwk);
  cvt_bf16<<<(nwq + 255) / 256, 256, 0, stream>>>(Wo, Wob, nwq);

  dim3 gq(E_DIM / 64, NTOK / 128);               // (16, 32)
  dim3 gk((NKVH * HDIM) / 64, NTOK / 128);       // (4, 32)
  gemm_bf16_wmma<<<gq, 256, 0, stream>>>(xb, Wqb, Qf, E_DIM, E_DIM);
  gemm_bf16_wmma<<<gk, 256, 0, stream>>>(xb, Wkb, Kf, E_DIM, NKVH * HDIM);
  gemm_bf16_wmma<<<gk, 256, 0, stream>>>(xb, Wvb, Vf, E_DIM, NKVH * HDIM);

  rope_cvt<<<(NTOK * NH * 32 + 255) / 256, 256, 0, stream>>>(Qf, cosb, sinb, Qr, NH, 1);
  rope_cvt<<<(NTOK * NKVH * 32 + 255) / 256, 256, 0, stream>>>(Kf, cosb, sinb, Kr, NKVH, 1);
  rope_cvt<<<(NTOK * NKVH * 32 + 255) / 256, 256, 0, stream>>>(Vf, cosb, sinb, Vb, NKVH, 0);

  dim3 gf(T_SEQ / 64, NH, 2);                    // (32, 16, 2)
  flash_gqa<<<gf, 128, 0, stream>>>(Qr, Kr, Vb, Yb);

  gemm_bf16_wmma<<<gq, 256, 0, stream>>>(Yb, Wob, (float*)d_out, E_DIM, E_DIM);
}
